// InfoNCELoss_88433376624844
// MI455X (gfx1250) — matrix operand order repaired and checked by reading, not compiled, not run on hardware
//
#include <hip/hip_runtime.h>

#define N_ROWS 8192
#define DIM 512
#define INV_T 14.285714285714286f /* 1/0.07 ; also the fixed logsumexp shift C */

typedef _Float16 v16h __attribute__((ext_vector_type(16)));
typedef _Float16 v8h  __attribute__((ext_vector_type(8)));
typedef _Float16 v4h  __attribute__((ext_vector_type(4)));
typedef float    v8f  __attribute__((ext_vector_type(8)));
typedef float    v4f  __attribute__((ext_vector_type(4)));

// ---------------------------------------------------------------------------
// Kernel 1: row-normalize and split each normalized fp32 value into f16 hi+lo
// (x = hi + lo to ~22 mantissa bits). One wave (32 lanes) per row of 512.
// ---------------------------------------------------------------------------
__global__ __launch_bounds__(256)
void normalize_split_kernel(const float* __restrict__ x,
                            _Float16* __restrict__ xh,
                            _Float16* __restrict__ xl) {
  const int wave = threadIdx.x >> 5;
  const int lane = threadIdx.x & 31;
  const int row  = blockIdx.x * 8 + wave;
  const v4f* xr4 = (const v4f*)(x + (size_t)row * DIM);

  v4f v[4];
  float ss = 0.0f;
#pragma unroll
  for (int i = 0; i < 4; ++i) {
    v[i] = xr4[i * 32 + lane];
    ss += v[i][0]*v[i][0] + v[i][1]*v[i][1] + v[i][2]*v[i][2] + v[i][3]*v[i][3];
  }
#pragma unroll
  for (int msk = 1; msk < 32; msk <<= 1) ss += __shfl_xor(ss, msk);

  const float scale = 1.0f / fmaxf(sqrtf(ss), 1e-8f);  // torch eps clamp
#pragma unroll
  for (int i = 0; i < 4; ++i) {
    v4h hi, lo;
#pragma unroll
    for (int j = 0; j < 4; ++j) {
      const float f = v[i][j] * scale;
      const _Float16 h = (_Float16)f;
      hi[j] = h;
      lo[j] = (_Float16)(f - (float)h);
    }
    const size_t off = (size_t)row * DIM + (size_t)(i * 32 + lane) * 4;
    *(v4h*)(xh + off) = hi;
    *(v4h*)(xl + off) = lo;
  }
}

// ---------------------------------------------------------------------------
// Kernel 2: fused Gram-GEMM + fixed-shift logsumexp.
// Block = 256 threads (8 waves) owns TWO 16-row tiles (32 rows, blockIdx.x).
// Both A tiles (32x512, hi+lo = 64 KB) staged in LDS; each wave sweeps 1024
// columns (64 16-col tiles). Per k-step: each B fragment (from L2) feeds
// 6x v_wmma_f32_16x16x32_f16 (hi*hi + hi*lo + lo*hi for both row tiles).
// Fixed logsumexp shift C = 1/T (max possible logit; diagonal contributes
// exp(0)=1 so the sum is >=1 -> no max pass needed).
// No atomics: LDS combine in fixed order -> partial[blockIdx.x].
// ---------------------------------------------------------------------------
__global__ __launch_bounds__(256, 2)
void gram_lse_kernel(const _Float16* __restrict__ xh,
                     const _Float16* __restrict__ xl,
                     float* __restrict__ partial) {
  __shared__ __align__(16) _Float16 ldsAh[32 * DIM];   // 32 KB
  __shared__ __align__(16) _Float16 ldsAl[32 * DIM];   // 32 KB
  __shared__ float ldsRowPart[8][32];
  __shared__ float ldsDiag[32];

  const int ib   = blockIdx.x;      // 32-row block (rows ib*32 .. +32)
  const int tid  = threadIdx.x;
  const int w    = tid >> 5;        // wave 0..7
  const int lane = tid & 31;
  const int m    = lane & 15;       // A-row / B-col within tile
  const int hh   = lane >> 4;       // half-wave select

  // ---- stage both A tiles (hi & lo) into LDS -----------------------------
  {
    const v8h* gh = (const v8h*)(xh + (size_t)ib * 32 * DIM);
    const v8h* gl = (const v8h*)(xl + (size_t)ib * 32 * DIM);
    v8h* sh = (v8h*)ldsAh;
    v8h* sl = (v8h*)ldsAl;
    for (int idx = tid; idx < 32 * DIM / 8; idx += 256) {
      sh[idx] = gh[idx];
      sl[idx] = gl[idx];
    }
  }
  __syncthreads();

  float s0[8], s1[8];
#pragma unroll
  for (int r = 0; r < 8; ++r) { s0[r] = 0.0f; s1[r] = 0.0f; }

  const _Float16* a0h_lds = ldsAh + m * DIM;          // tile0 rows 0..15
  const _Float16* a0l_lds = ldsAl + m * DIM;
  const _Float16* a1h_lds = ldsAh + (m + 16) * DIM;   // tile1 rows 16..31
  const _Float16* a1l_lds = ldsAl + (m + 16) * DIM;

  const int dct0 = ib * 2;          // diagonal column tile for rows 0..15
  const int dct1 = ib * 2 + 1;      // diagonal column tile for rows 16..31

  for (int t = 0; t < 64; ++t) {
    const int gct = (w << 6) + t;                       // global column tile
    const _Float16* bh_row = xh + (size_t)((gct << 4) + m) * DIM;
    const _Float16* bl_row = xl + (size_t)((gct << 4) + m) * DIM;

    if (t + 1 < 64) {  // prefetch next tile's B rows into caches
      __builtin_prefetch((const void*)(bh_row + (16 * DIM)), 0, 1);
      __builtin_prefetch((const void*)(bl_row + (16 * DIM)), 0, 1);
    }

    v8f c0 = {};
    v8f c1 = {};
#pragma unroll 4
    for (int kk = 0; kk < 16; ++kk) {
      const int kb = kk << 5;
      // A fragment: halves [0..7]=K[kb+hh*8..], [8..15]=K[kb+16+hh*8..]
      v8h t00 = *(const v8h*)(a0h_lds + kb + hh * 8);
      v8h t01 = *(const v8h*)(a0h_lds + kb + 16 + hh * 8);
      v8h t02 = *(const v8h*)(a0l_lds + kb + hh * 8);
      v8h t03 = *(const v8h*)(a0l_lds + kb + 16 + hh * 8);
      v8h t10 = *(const v8h*)(a1h_lds + kb + hh * 8);
      v8h t11 = *(const v8h*)(a1h_lds + kb + 16 + hh * 8);
      v8h t12 = *(const v8h*)(a1l_lds + kb + hh * 8);
      v8h t13 = *(const v8h*)(a1l_lds + kb + 16 + hh * 8);
      v16h a0h = __builtin_shufflevector(t00, t01, 0,1,2,3,4,5,6,7,8,9,10,11,12,13,14,15);
      v16h a0l = __builtin_shufflevector(t02, t03, 0,1,2,3,4,5,6,7,8,9,10,11,12,13,14,15);
      v16h a1h = __builtin_shufflevector(t10, t11, 0,1,2,3,4,5,6,7,8,9,10,11,12,13,14,15);
      v16h a1l = __builtin_shufflevector(t12, t13, 0,1,2,3,4,5,6,7,8,9,10,11,12,13,14,15);
      // B fragment: halves [0..7]=K[kb+hh*8..], [8..15]=K[kb+16+hh*8..] of column (gct*16 + m)
      v16h bh = __builtin_shufflevector(*(const v8h*)(bh_row + kb + hh * 8), *(const v8h*)(bh_row + kb + 16 + hh * 8), 0,1,2,3,4,5,6,7,8,9,10,11,12,13,14,15);
      v16h bl = __builtin_shufflevector(*(const v8h*)(bl_row + kb + hh * 8), *(const v8h*)(bl_row + kb + 16 + hh * 8), 0,1,2,3,4,5,6,7,8,9,10,11,12,13,14,15);

      c0 = __builtin_amdgcn_wmma_f32_16x16x32_f16(false, a0h, false, bh, (short)0, c0, false, false);
      c0 = __builtin_amdgcn_wmma_f32_16x16x32_f16(false, a0h, false, bl, (short)0, c0, false, false);
      c0 = __builtin_amdgcn_wmma_f32_16x16x32_f16(false, a0l, false, bh, (short)0, c0, false, false);
      c1 = __builtin_amdgcn_wmma_f32_16x16x32_f16(false, a1h, false, bh, (short)0, c1, false, false);
      c1 = __builtin_amdgcn_wmma_f32_16x16x32_f16(false, a1h, false, bl, (short)0, c1, false, false);
      c1 = __builtin_amdgcn_wmma_f32_16x16x32_f16(false, a1l, false, bh, (short)0, c1, false, false);
    }

    const bool isDiag0 = (gct == dct0);
    const bool isDiag1 = (gct == dct1);
#pragma unroll
    for (int r = 0; r < 8; ++r) {
      const float l0 = c0[r] * INV_T;          // row = r+8*hh,    col = gct*16+m
      const float l1 = c1[r] * INV_T;          // row = 16+r+8*hh, col = gct*16+m
      s0[r] += __expf(l0 - INV_T);             // fixed shift C = 1/T
      s1[r] += __expf(l1 - INV_T);
      if (isDiag0 && m == r + 8 * hh) ldsDiag[r + 8 * hh] = l0;
      if (isDiag1 && m == r + 8 * hh) ldsDiag[16 + r + 8 * hh] = l1;
    }
  }

  // combine the 16 column-lanes per row (stays within half on wave32)
#pragma unroll
  for (int r = 0; r < 8; ++r) {
    float v0 = s0[r];
    float v1 = s1[r];
    v0 += __shfl_xor(v0, 1); v1 += __shfl_xor(v1, 1);
    v0 += __shfl_xor(v0, 2); v1 += __shfl_xor(v1, 2);
    v0 += __shfl_xor(v0, 4); v1 += __shfl_xor(v1, 4);
    v0 += __shfl_xor(v0, 8); v1 += __shfl_xor(v1, 8);
    if (m == 0) {
      ldsRowPart[w][r + 8 * hh]      = v0;
      ldsRowPart[w][16 + r + 8 * hh] = v1;
    }
  }
  __syncthreads();

  if (tid < 32) {
    float tot = 0.0f;
#pragma unroll
    for (int ww = 0; ww < 8; ++ww) tot += ldsRowPart[ww][tid];  // fixed order
    float contrib = INV_T + __logf(tot) - ldsDiag[tid];         // lse - diag
    contrib += __shfl_xor(contrib, 1);
    contrib += __shfl_xor(contrib, 2);
    contrib += __shfl_xor(contrib, 4);
    contrib += __shfl_xor(contrib, 8);
    contrib += __shfl_xor(contrib, 16);
    if (tid == 0) partial[ib] = contrib;       // sum over this block's 32 rows
  }
}

// ---------------------------------------------------------------------------
// Kernel 3: deterministic reduction of 256 block partials -> mean
// ---------------------------------------------------------------------------
__global__ __launch_bounds__(256)
void final_reduce_kernel(const float* __restrict__ partial, float* __restrict__ out) {
  __shared__ float buf[256];
  buf[threadIdx.x] = partial[threadIdx.x];
  __syncthreads();
  for (int sft = 128; sft > 0; sft >>= 1) {
    if ((int)threadIdx.x < sft) buf[threadIdx.x] += buf[threadIdx.x + sft];
    __syncthreads();
  }
  if (threadIdx.x == 0) out[0] = buf[0] * (1.0f / (float)N_ROWS);
}

extern "C" void kernel_launch(void* const* d_in, const int* in_sizes, int n_in,
                              void* d_out, int out_size, void* d_ws, size_t ws_size,
                              hipStream_t stream) {
  (void)in_sizes; (void)n_in; (void)out_size; (void)ws_size;
  const float* x = (const float*)d_in[0];          // [512,16,512] fp32
  float* out = (float*)d_out;                      // scalar fp32

  // workspace layout: xh (8 MB) | xl (8 MB) | partial (256 f32)
  _Float16* xh = (_Float16*)d_ws;
  _Float16* xl = xh + (size_t)N_ROWS * DIM;
  float* partial = (float*)(xl + (size_t)N_ROWS * DIM);

  normalize_split_kernel<<<N_ROWS / 8, 256, 0, stream>>>(x, xh, xl);
  gram_lse_kernel<<<N_ROWS / 32, 256, 0, stream>>>(xh, xl, partial);
  final_reduce_kernel<<<1, 256, 0, stream>>>(partial, out);
}
